// Clustering_20753281974691
// MI455X (gfx1250) — compile-verified, matrix operand order checked by reading, not run on hardware
//
#include <hip/hip_runtime.h>
#include <hip/hip_bf16.h>

// ---------------------------------------------------------------------------
// CDNA5 (gfx1250) implementation of the clustering-attention block.
// All GEMMs run on v_wmma_f32_16x16x32_bf16 (bf16 inputs, f32 accumulate),
// register-blocked so the inner loops sustain ~2 b128 loads per WMMA.
// ---------------------------------------------------------------------------

typedef __bf16 bf16_t;
typedef __attribute__((ext_vector_type(8)))  bf16_t v8bf;
typedef __attribute__((ext_vector_type(16))) bf16_t v16bf;
typedef __attribute__((ext_vector_type(8)))  float  v8f;

union FragAB { v16bf v; v8bf h[2]; };

// Problem constants
#define BATCH   8
#define CDIM    512
#define HEADS   8
#define HD      64
#define NTOK    3600           // 60*60
#define NTOKP   3616           // padded to multiple of 32 (K-dim use)
#define MROWS   28800          // 8*3600
#define MCL     100            // clusters
#define MCLP    112            // padded to multiple of 16
#define BH      64             // BATCH*HEADS
#define QKVN    1536

// Fragment loader: row-major tile, two contiguous 16B runs per lane.
// Matches CDNA5 16-bit A layout (lanes 0-15: K 0..7 & 16..23; lanes 16-31: +8).
// For B, pass the N-major (transposed) table: its rows are B's columns.
__device__ __forceinline__ v16bf load_frag(const bf16_t* __restrict__ tile, int ld) {
    const int lane = threadIdx.x & 31;
    const int row  = lane & 15;
    const int koff = (lane >> 4) << 3;           // 0 or 8
    const bf16_t* p = tile + row * ld + koff;
    FragAB f;
    f.h[0] = *(const v8bf*)(p);
    f.h[1] = *(const v8bf*)(p + 16);
    return f.v;
}

__device__ __forceinline__ v8f wmma_bf16(v16bf a, v16bf b, v8f c) {
    return __builtin_amdgcn_wmma_f32_16x16x32_bf16(false, a, false, b, (short)0, c,
                                                   false, false);
}

// ------------------------------ small utility kernels ----------------------

// transpose f32 (K x N) -> bf16 (N x K)
__global__ void k_transpose_w(const float* __restrict__ w, bf16_t* __restrict__ wt,
                              int K, int N) {
    int i = blockIdx.x * blockDim.x + threadIdx.x;
    if (i >= K * N) return;
    int n = i % N, k = i / N;
    wt[n * K + k] = (bf16_t)w[k * N + n];
}

__global__ void k_f32_to_bf16(const float* __restrict__ in, bf16_t* __restrict__ out,
                              int nelem) {
    int i = blockIdx.x * blockDim.x + threadIdx.x;
    if (i < nelem) out[i] = (bf16_t)in[i];
}

__global__ void k_zero_i32(int* __restrict__ p, int nelem) {
    int i = blockIdx.x * blockDim.x + threadIdx.x;
    if (i < nelem) p[i] = 0;
}

// zero the 16 padded token columns of f_tT [BH][HD][NTOKP]
__global__ void k_pad_ftT(bf16_t* __restrict__ ftT) {
    int i = blockIdx.x * blockDim.x + threadIdx.x;   // BH*HD*16
    if (i >= BH * HD * 16) return;
    int c = i & 15, row = i >> 4;
    ftT[row * NTOKP + NTOK + c] = (bf16_t)0.0f;
}

// ------------------------------ QKV GEMM -----------------------------------
// C(28800x1536) = X_bf16 @ W^T + b, scattered directly into token tables.
// One wave computes a 32x32 output block: 8 b128 loads / 4 WMMA per K-step.
__global__ void k_gemm_qkv(const bf16_t* __restrict__ X,   // [28800][512]
                           const bf16_t* __restrict__ WT,  // [1536][512]
                           const float*  __restrict__ bias,
                           float*  __restrict__ qt,        // [BH][3600][64] f32
                           bf16_t* __restrict__ vt,        // [BH][3600][64] bf16
                           float*  __restrict__ ft,        // [BH][3600][64] f32
                           bf16_t* __restrict__ ftT) {     // [BH][64][3616] bf16
    const int wave   = threadIdx.x >> 5;
    const int ntile2 = blockIdx.x * 4 + wave;    // 0..47 (32-col blocks)
    const int mtile2 = blockIdx.y;               // 0..899 (32-row blocks)
    const bf16_t* At = X  + mtile2 * 32 * CDIM;
    const bf16_t* Bt = WT + ntile2 * 32 * CDIM;
    v8f c00 = {}, c01 = {}, c10 = {}, c11 = {};
    for (int k0 = 0; k0 < CDIM; k0 += 32) {
        __builtin_prefetch(At + k0 + 64, 0, 1);
        __builtin_prefetch(Bt + k0 + 64, 0, 1);
        v16bf a0 = load_frag(At + k0, CDIM);
        v16bf a1 = load_frag(At + 16 * CDIM + k0, CDIM);
        v16bf b0 = load_frag(Bt + k0, CDIM);
        v16bf b1 = load_frag(Bt + 16 * CDIM + k0, CDIM);
        c00 = wmma_bf16(a0, b0, c00);
        c01 = wmma_bf16(a0, b1, c01);
        c10 = wmma_bf16(a1, b0, c10);
        c11 = wmma_bf16(a1, b1, c11);
    }
    const int lane  = threadIdx.x & 31;
    const int col   = lane & 15;
    const int rbase = (lane >> 4) << 3;
    v8f acc[2][2] = {{c00, c01}, {c10, c11}};
#pragma unroll
    for (int ti = 0; ti < 2; ++ti) {
#pragma unroll
        for (int tj = 0; tj < 2; ++tj) {
            const int n     = ntile2 * 32 + tj * 16 + col;
            const int which = n >> 9;       // 0=q 1=v 2=f
            const int head  = (n >> 6) & 7;
            const int d     = n & 63;
            const float bv  = bias[n];
#pragma unroll
            for (int r = 0; r < 8; ++r) {
                int m   = mtile2 * 32 + ti * 16 + rbase + r;
                int bb  = m / NTOK, pix = m % NTOK;
                int bh  = bb * HEADS + head;
                int tok = (bh * NTOK + pix) * HD + d;
                float val = acc[ti][tj][r] + bv;
                if (which == 0)      qt[tok] = val;
                else if (which == 1) vt[tok] = (bf16_t)val;
                else { ft[tok] = val; ftT[(bh * HD + d) * NTOKP + pix] = (bf16_t)val; }
            }
        }
    }
}

// ------------------------------ pooling ------------------------------------
// centers = avgpool6x6(q), centers_feature = avgpool6x6(f); rows >=100 zeroed.
__global__ void k_pool(const float* __restrict__ qt, const float* __restrict__ ft,
                       float* __restrict__ centers, bf16_t* __restrict__ centers_b,
                       float* __restrict__ centers_feat) {
    int bm = blockIdx.x;                  // bh*112 + m
    int bh = bm / MCLP, m = bm % MCLP;
    int d  = threadIdx.x;                 // 0..63
    float sq = 0.0f, sf = 0.0f;
    if (m < MCL) {
        int mh = m / 10, mw = m % 10;
        for (int r = 0; r < 6; ++r)
            for (int cc = 0; cc < 6; ++cc) {
                int pix = (mh * 6 + r) * 60 + (mw * 6 + cc);
                int idx = (bh * NTOK + pix) * HD + d;
                sq += qt[idx];
                sf += ft[idx];
            }
        sq *= (1.0f / 36.0f);
        sf *= (1.0f / 36.0f);
    }
    int o = bm * HD + d;
    centers[o]      = sq;
    centers_b[o]    = (bf16_t)sq;
    centers_feat[o] = sf;
}

// ------------------------------ attn / cos GEMM ----------------------------
// out[bh][m][n] = sum_d A[bh][m][d] * T[bh][n][d]   (M=112, N=3600, K=64)
// A-fragments are loop-invariant per wave; each wave sweeps 15 N-tiles.
__global__ void k_attn_gemm(const bf16_t* __restrict__ A,  // [BH][112][64]
                            const bf16_t* __restrict__ T,  // [BH][3600][64]
                            float* __restrict__ out) {     // [BH][112][3616]
    const int mtile = threadIdx.x >> 5;   // 0..6
    const int bh    = blockIdx.y;
    const bf16_t* At = A + (bh * MCLP + mtile * 16) * HD;
    const v16bf a0 = load_frag(At + 0,  HD);
    const v16bf a1 = load_frag(At + 32, HD);
    const int lane  = threadIdx.x & 31;
    const int col   = lane & 15;
    const int rbase = (lane >> 4) << 3;
    for (int j = 0; j < 15; ++j) {
        const int ntile = blockIdx.x * 15 + j;        // 0..224
        const bf16_t* Bt = T + (bh * NTOK + ntile * 16) * HD;
        v16bf b0 = load_frag(Bt + 0,  HD);
        v16bf b1 = load_frag(Bt + 32, HD);
        v8f c = {};
        c = wmma_bf16(a0, b0, c);
        c = wmma_bf16(a1, b1, c);
        const int n = ntile * 16 + col;
#pragma unroll
        for (int r = 0; r < 8; ++r) {
            int m = mtile * 16 + rbase + r;
            out[(bh * MCLP + m) * NTOKP + n] = c[r];
        }
    }
}

// ------------------------------ softmax over clusters ----------------------
__global__ void k_softmax(const float* __restrict__ attn, bf16_t* __restrict__ attnb) {
    int n  = blockIdx.x * blockDim.x + threadIdx.x;
    int bh = blockIdx.y;
    if (n >= NTOKP) return;
    int base = bh * MCLP * NTOKP + n;
    if (n < NTOK) {
        float mx = -3.4e38f;
        for (int m = 0; m < MCL; ++m) mx = fmaxf(mx, attn[base + m * NTOKP]);
        float s = 0.0f;
        for (int m = 0; m < MCL; ++m) s += __expf(attn[base + m * NTOKP] - mx);
        float inv = 1.0f / s;
        for (int m = 0; m < MCL; ++m)
            attnb[base + m * NTOKP] = (bf16_t)(__expf(attn[base + m * NTOKP] - mx) * inv);
        for (int m = MCL; m < MCLP; ++m) attnb[base + m * NTOKP] = (bf16_t)0.0f;
    } else {
        for (int m = 0; m < MCLP; ++m) attnb[base + m * NTOKP] = (bf16_t)0.0f;
    }
}

// ------------------------------ update / agg GEMM --------------------------
// outf[bh][m][d] = sum_n A[bh][m][n] * f_t[n][d]   (M=112, N=64, K=3616)
// One wave computes a full 16x64 row block: A loaded once per K-step,
// reused across 4 B-tiles (10 loads / 4 WMMA).
__global__ void k_update_gemm(const bf16_t* __restrict__ A,   // [BH][112][3616]
                              const bf16_t* __restrict__ BT,  // [BH][64][3616]
                              float*  __restrict__ outf,      // [BH][112][64]
                              bf16_t* __restrict__ outb) {    // optional bf16 copy
    const int mtile = blockIdx.x;          // 0..6
    const int bh    = blockIdx.y;
    const bf16_t* At = A  + (bh * MCLP + mtile * 16) * NTOKP;
    const bf16_t* Bt = BT + bh * HD * NTOKP;
    v8f c0 = {}, c1 = {}, c2 = {}, c3 = {};
    for (int k0 = 0; k0 < NTOKP; k0 += 32) {
        __builtin_prefetch(At + k0 + 64, 0, 1);
        v16bf a  = load_frag(At + k0, NTOKP);
        v16bf b0 = load_frag(Bt + 0  * 16 * NTOKP + k0, NTOKP);
        v16bf b1 = load_frag(Bt + 1  * 16 * NTOKP + k0, NTOKP);
        v16bf b2 = load_frag(Bt + 2  * 16 * NTOKP + k0, NTOKP);
        v16bf b3 = load_frag(Bt + 3  * 16 * NTOKP + k0, NTOKP);
        c0 = wmma_bf16(a, b0, c0);
        c1 = wmma_bf16(a, b1, c1);
        c2 = wmma_bf16(a, b2, c2);
        c3 = wmma_bf16(a, b3, c3);
    }
    const int lane  = threadIdx.x & 31;
    const int col   = lane & 15;
    const int rbase = (lane >> 4) << 3;
    v8f acc[4] = {c0, c1, c2, c3};
#pragma unroll
    for (int j = 0; j < 4; ++j) {
#pragma unroll
        for (int r = 0; r < 8; ++r) {
            int m = mtile * 16 + rbase + r;
            int o = (bh * MCLP + m) * HD + j * 16 + col;
            outf[o] = acc[j][r];
            if (outb) outb[o] = (bf16_t)acc[j][r];
        }
    }
}

// ------------------------------ row L2-normalize (width 64) ----------------
__global__ void k_norm_rows(const float* __restrict__ in, bf16_t* __restrict__ out,
                            int rows) {
    int r = blockIdx.x * blockDim.x + threadIdx.x;
    if (r >= rows) return;
    const float* p = in + r * HD;
    float s = 0.0f;
    for (int d = 0; d < HD; ++d) s += p[d] * p[d];
    float inv = 1.0f / fmaxf(sqrtf(s), 1e-12f);
    bf16_t* o = out + r * HD;
    for (int d = 0; d < HD; ++d) o[d] = (bf16_t)(p[d] * inv);
}

// ------------------------------ argmax + mask ------------------------------
__global__ void k_argmax(const float* __restrict__ cosb,     // [BH][112][3616]
                         const float* __restrict__ alpha, const float* __restrict__ beta,
                         bf16_t* __restrict__ simb,          // [BH][112][3616]
                         int* __restrict__ idxv, float* __restrict__ sval,
                         int* __restrict__ counts) {
    int n  = blockIdx.x * blockDim.x + threadIdx.x;
    int bh = blockIdx.y;
    if (n >= NTOKP) return;
    int base = bh * MCLP * NTOKP + n;
    if (n < NTOK) {
        float al = alpha[0], be = beta[0];
        float best = -3.4e38f; int bi = 0;
        for (int m = 0; m < MCL; ++m) {
            float t = al * cosb[base + m * NTOKP];   // sigmoid is monotone: argmax ok
            if (t > best) { best = t; bi = m; }
        }
        float s = 1.0f / (1.0f + __expf(-(be + best)));
        idxv[bh * NTOK + n] = bi;
        sval[bh * NTOK + n] = s;
        atomicAdd(&counts[bh * MCLP + bi], 1);
        for (int m = 0; m < MCLP; ++m)
            simb[base + m * NTOKP] = (m == bi) ? (bf16_t)s : (bf16_t)0.0f;
    } else {
        for (int m = 0; m < MCLP; ++m) simb[base + m * NTOKP] = (bf16_t)0.0f;
    }
}

// ------------------------------ out_c and token scatter --------------------
__global__ void k_out_c(const float* __restrict__ agg, const float* __restrict__ cf,
                        const int* __restrict__ counts, float* __restrict__ outc) {
    int bm = blockIdx.x, d = threadIdx.x;
    int o  = bm * HD + d;
    outc[o] = (agg[o] + cf[o]) / ((float)counts[bm] + 1.0f);
}

// out_t[n] = out_c[argmax[n]] * sim[argmax[n], n]  (single surviving cluster)
__global__ void k_out_scatter(const float* __restrict__ outc, const int* __restrict__ idxv,
                              const float* __restrict__ sval, bf16_t* __restrict__ OUT) {
    int t  = blockIdx.x;                 // bh*3600 + n
    int d  = threadIdx.x;
    int bh = t / NTOK, n = t % NTOK;
    int bb = bh >> 3, head = bh & 7;
    float v = outc[(bh * MCLP + idxv[t]) * HD + d] * sval[t];
    OUT[(bb * NTOK + n) * CDIM + head * HD + d] = (bf16_t)v;
}

// ------------------------------ projection GEMM ----------------------------
// 32x32 per wave, like the QKV GEMM.
__global__ void k_gemm_proj(const bf16_t* __restrict__ A,   // [28800][512]
                            const bf16_t* __restrict__ WT,  // [512][512]
                            const float*  __restrict__ bias,
                            float* __restrict__ Y) {        // [28800][512]
    const int wave   = threadIdx.x >> 5;
    const int ntile2 = blockIdx.x * 4 + wave;  // 0..15
    const int mtile2 = blockIdx.y;             // 0..899
    const bf16_t* At = A  + mtile2 * 32 * CDIM;
    const bf16_t* Bt = WT + ntile2 * 32 * CDIM;
    v8f c00 = {}, c01 = {}, c10 = {}, c11 = {};
    for (int k0 = 0; k0 < CDIM; k0 += 32) {
        __builtin_prefetch(At + k0 + 64, 0, 1);
        __builtin_prefetch(Bt + k0 + 64, 0, 1);
        v16bf a0 = load_frag(At + k0, CDIM);
        v16bf a1 = load_frag(At + 16 * CDIM + k0, CDIM);
        v16bf b0 = load_frag(Bt + k0, CDIM);
        v16bf b1 = load_frag(Bt + 16 * CDIM + k0, CDIM);
        c00 = wmma_bf16(a0, b0, c00);
        c01 = wmma_bf16(a0, b1, c01);
        c10 = wmma_bf16(a1, b0, c10);
        c11 = wmma_bf16(a1, b1, c11);
    }
    const int lane  = threadIdx.x & 31;
    const int col   = lane & 15;
    const int rbase = (lane >> 4) << 3;
    v8f acc[2][2] = {{c00, c01}, {c10, c11}};
#pragma unroll
    for (int ti = 0; ti < 2; ++ti) {
#pragma unroll
        for (int tj = 0; tj < 2; ++tj) {
            const int n    = ntile2 * 32 + tj * 16 + col;
            const float bv = bias[n];
#pragma unroll
            for (int r = 0; r < 8; ++r) {
                int m = mtile2 * 32 + ti * 16 + rbase + r;
                Y[m * CDIM + n] = acc[ti][tj][r] + bv;
            }
        }
    }
}

// ------------------------------ host launcher ------------------------------

static inline size_t align256(size_t x) { return (x + 255) & ~(size_t)255; }

extern "C" void kernel_launch(void* const* d_in, const int* in_sizes, int n_in,
                              void* d_out, int out_size, void* d_ws, size_t ws_size,
                              hipStream_t stream) {
    const float* x       = (const float*)d_in[0];   // (8,60,60,512)
    const float* qkv_w   = (const float*)d_in[1];   // (512,1536)
    const float* qkv_b   = (const float*)d_in[2];   // (1536,)
    const float* proj_w  = (const float*)d_in[3];   // (512,512)
    const float* proj_b  = (const float*)d_in[4];   // (512,)
    const float* sim_al  = (const float*)d_in[5];
    const float* sim_be  = (const float*)d_in[6];
    float* Y             = (float*)d_out;

    // workspace layout
    char* ws = (char*)d_ws;
    size_t off = 0;
    auto take = [&](size_t bytes) { char* p = ws + off; off += align256(bytes); return p; };

    bf16_t* WT       = (bf16_t*)take((size_t)QKVN * CDIM * 2);       // 1.5 MB
    bf16_t* PWT      = (bf16_t*)take((size_t)CDIM * CDIM * 2);       // 0.5 MB
    bf16_t* x_bf     = (bf16_t*)take((size_t)MROWS * CDIM * 2);      // 29.5 MB (reused as q_tn)
    float*  q_t      = (float*) take((size_t)BH * NTOK * HD * 4);    // 59 MB (reused as OUT bf16)
    float*  f_t      = (float*) take((size_t)BH * NTOK * HD * 4);    // 59 MB
    bf16_t* v_t      = (bf16_t*)take((size_t)BH * NTOK * HD * 2);    // 29.5 MB
    bf16_t* ftT      = (bf16_t*)take((size_t)BH * HD * NTOKP * 2);   // 29.6 MB
    float*  attn     = (float*) take((size_t)BH * MCLP * NTOKP * 4); // 103.7 MB
    bf16_t* attnb    = (bf16_t*)take((size_t)BH * MCLP * NTOKP * 2); // 51.8 MB (reused as sim)
    float*  centers  = (float*) take((size_t)BH * MCLP * HD * 4);
    bf16_t* centersb = (bf16_t*)take((size_t)BH * MCLP * HD * 2);
    float*  cfeat    = (float*) take((size_t)BH * MCLP * HD * 4);
    bf16_t* centersn = (bf16_t*)take((size_t)BH * MCLP * HD * 2);
    float*  agg      = (float*) take((size_t)BH * MCLP * HD * 4);
    float*  outc     = (float*) take((size_t)BH * MCLP * HD * 4);
    int*    counts   = (int*)   take((size_t)BH * MCLP * 4);
    int*    idxv     = (int*)   take((size_t)BH * NTOK * 4);
    float*  sval     = (float*) take((size_t)BH * NTOK * 4);

    bf16_t* q_tn   = x_bf;            // reuse (same size, lifetime disjoint)
    bf16_t* simb   = attnb;           // reuse
    bf16_t* OUTbf  = (bf16_t*)q_t;    // reuse (smaller, lifetime disjoint)
    (void)ws_size; (void)in_sizes; (void)n_in; (void)out_size;

    // 1) weight transposes + input conversion
    k_transpose_w<<<(CDIM * QKVN + 255) / 256, 256, 0, stream>>>(qkv_w, WT, CDIM, QKVN);
    k_transpose_w<<<(CDIM * CDIM + 255) / 256, 256, 0, stream>>>(proj_w, PWT, CDIM, CDIM);
    k_f32_to_bf16<<<(MROWS * CDIM + 255) / 256, 256, 0, stream>>>(x, x_bf, MROWS * CDIM);

    // 2) QKV GEMM with scatter into token tables
    {
        dim3 grid(QKVN / 128, MROWS / 32);   // 12 x 900, 4 waves/block, 32x32/wave
        k_gemm_qkv<<<grid, 128, 0, stream>>>(x_bf, WT, qkv_b, q_t, v_t, f_t, ftT);
    }
    k_pad_ftT<<<(BH * HD * 16 + 255) / 256, 256, 0, stream>>>(ftT);
    k_zero_i32<<<(BH * MCLP + 255) / 256, 256, 0, stream>>>(counts, BH * MCLP);

    // 3) average-pool initial centers
    k_pool<<<BH * MCLP, HD, 0, stream>>>(q_t, f_t, centers, centersb, cfeat);

    // 4) clustering iterations
    for (int it = 0; it < 3; ++it) {
        dim3 g1(15, BH);                    // 15 blocks x 15 n-tiles, 7 waves/block
        k_attn_gemm<<<g1, 224, 0, stream>>>(centersb, v_t, attn);
        dim3 g2((NTOKP + 255) / 256, BH);
        k_softmax<<<g2, 256, 0, stream>>>(attn, attnb);
        dim3 g3(MCLP / 16, BH);             // 7 x 64, one wave per 16x64 block
        k_update_gemm<<<g3, 32, 0, stream>>>(attnb, ftT, centers, centersb);
    }

    // 5) cosine similarity
    k_norm_rows<<<(BH * MCLP + 255) / 256, 256, 0, stream>>>(centers, centersn, BH * MCLP);
    k_norm_rows<<<(BH * NTOK + 255) / 256, 256, 0, stream>>>(q_t, q_tn, BH * NTOK);
    {
        dim3 g1(15, BH);
        k_attn_gemm<<<g1, 224, 0, stream>>>(centersn, q_tn, attn);   // attn <- cos
    }

    // 6) argmax / mask / aggregation
    {
        dim3 g2((NTOKP + 255) / 256, BH);
        k_argmax<<<g2, 256, 0, stream>>>(attn, sim_al, sim_be, simb, idxv, sval, counts);
        dim3 g3(MCLP / 16, BH);
        k_update_gemm<<<g3, 32, 0, stream>>>(simb, ftT, agg, (bf16_t*)nullptr);
    }
    k_out_c<<<BH * MCLP, HD, 0, stream>>>(agg, cfeat, counts, outc);
    k_out_scatter<<<BH * NTOK, HD, 0, stream>>>(outc, idxv, sval, OUTbf);

    // 7) output projection
    {
        dim3 grid(CDIM / 128, MROWS / 32);   // 4 x 900
        k_gemm_proj<<<grid, 128, 0, stream>>>(OUTbf, PWT, proj_b, Y);
    }
}
